// GCNWithAttentionOneHead_13469017441154
// MI455X (gfx1250) — matrix operand order repaired and checked by reading, not compiled, not run on hardware
//
#include <hip/hip_runtime.h>
#include <hip/hip_bf16.h>

#define NN 10000
#define KK 32
#define DD 128
#define HH 256
#define LDP 264   // padded LDS row stride (bf16 elems) for activations: conflict-free b128
#define WLP 264   // padded LDS row stride (bf16 elems) for weights
#define GRID 1024

typedef __attribute__((ext_vector_type(16))) __bf16 bf16x16;
typedef __attribute__((ext_vector_type(8)))  __bf16 bf16x8;
typedef __attribute__((ext_vector_type(8)))  float  f32x8;
typedef __attribute__((ext_vector_type(4)))  float  f32x4;

// Dynamic LDS budget: W1 + W2 (256x264 bf16 each) + zA + zB (32x264 bf16 each)
//                     + b1,b2,W3 (256 f32 each) + mlp (32 f32) + nbr (32 i32)
#define SMEM_BYTES ((2 * HH * WLP + 2 * KK * LDP) * 2 + (3 * HH + KK) * 4 + KK * 4)
// = 304128 + 3072 + 128 + 128 = 307456 B  (<= 320KB per-WGP LDS)

// ---------------- prep: transpose W1/W2 (256x256 f32, row-major [k][n]) into bf16 W^T [n][k]
__global__ void prep_weights(const float* __restrict__ W1,
                             const float* __restrict__ W2,
                             __bf16* __restrict__ W1t,
                             __bf16* __restrict__ W2t) {
  int e = blockIdx.x * blockDim.x + threadIdx.x;   // 0..65535
  int k = e >> 8;
  int n = e & 255;
  W1t[(size_t)n * HH + k] = (__bf16)W1[e];
  W2t[(size_t)n * HH + k] = (__bf16)W2[e];
}

// ---------------- zero-fill the 10000x10000 output matrix (scatter target)
__global__ void zero_out(float* __restrict__ pw, size_t count4) {
  size_t i = (size_t)blockIdx.x * blockDim.x + threadIdx.x;
  size_t stride = (size_t)gridDim.x * blockDim.x;
  f32x4* p = (f32x4*)pw;
  f32x4 z = {0.f, 0.f, 0.f, 0.f};
  for (size_t j = i; j < count4; j += stride) p[j] = z;
}

// ---------------- one dense layer: zout = relu(zin @ W + bias), 32x256 @ 256x256
// zin/zout: LDS bf16 (stride LDP). WL: LDS bf16, transposed [n][k] (stride WLP). biasL: LDS f32.
__device__ __forceinline__ void dense_layer_wmma(const __bf16* zin, __bf16* zout,
                                                 const __bf16* WL,
                                                 const float* biasL,
                                                 int lane, int wave) {
  const int lh = lane & 15;
  const int lg = lane >> 4;   // half-wave select

  f32x8 acc[2][2];
#pragma unroll
  for (int mt = 0; mt < 2; ++mt)
#pragma unroll
    for (int nt = 0; nt < 2; ++nt)
#pragma unroll
      for (int r = 0; r < 8; ++r) acc[mt][nt][r] = 0.0f;

#pragma unroll
  for (int kk = 0; kk < 8; ++kk) {
    // A fragments (16x32 bf16): lanes 0-15 hold K {0..7,16..23}, lanes 16-31 hold {8..15,24..31}
    bf16x16 afrag[2];
    const int kbA = kk * 32 + lg * 8;
#pragma unroll
    for (int mt = 0; mt < 2; ++mt) {
      const __bf16* pa = zin + (size_t)(mt * 16 + lh) * LDP + kbA;
      bf16x8 lo = *(const bf16x8*)pa;          // ds_load_b128
      bf16x8 hi = *(const bf16x8*)(pa + 16);   // ds_load_b128
      afrag[mt] = __builtin_shufflevector(lo, hi, 0, 1, 2, 3, 4, 5, 6, 7,
                                          8, 9, 10, 11, 12, 13, 14, 15);
    }
    // B fragments (32x16 bf16): lane = column, lanes 0-15 K {0..7,16..23}, lanes 16-31 K {8..15,24..31}.
    // W is LDS-resident and transposed, so a fragment is two 8-bf16 runs at k and k+16 (two ds_load_b128).
    bf16x16 bfrag[2];
    const int kbB = kk * 32 + lg * 8;
#pragma unroll
    for (int nt = 0; nt < 2; ++nt) {
      const int col = (wave * 2 + nt) * 16 + lh;
      const __bf16* pb = WL + (size_t)col * WLP + kbB;
      bf16x8 lo = *(const bf16x8*)pb;
      bf16x8 hi = *(const bf16x8*)(pb + 16);
      bfrag[nt] = __builtin_shufflevector(lo, hi, 0, 1, 2, 3, 4, 5, 6, 7,
                                          8, 9, 10, 11, 12, 13, 14, 15);
    }
#pragma unroll
    for (int mt = 0; mt < 2; ++mt)
#pragma unroll
      for (int nt = 0; nt < 2; ++nt)
        acc[mt][nt] = __builtin_amdgcn_wmma_f32_16x16x32_bf16(
            false, afrag[mt], false, bfrag[nt], (short)0, acc[mt][nt],
            false, false);
  }

  // bias + relu + store back to LDS (C layout: VGPR r -> row r + 8*lg, col = lane&15)
#pragma unroll
  for (int mt = 0; mt < 2; ++mt)
#pragma unroll
    for (int nt = 0; nt < 2; ++nt) {
      const int col = (wave * 2 + nt) * 16 + lh;
      const float bv = biasL[col];
#pragma unroll
      for (int r = 0; r < 8; ++r) {
        const int row = mt * 16 + lg * 8 + r;
        float v = acc[mt][nt][r] + bv;
        zout[(size_t)row * LDP + col] = (__bf16)fmaxf(v, 0.0f);
      }
    }
}

// ---------------- main: persistent workgroups; weights live in LDS across nodes
__global__ __launch_bounds__(256) void gcn_attn_kernel(
    const float* __restrict__ x, const int* __restrict__ nbrs,
    const float* __restrict__ t, const float* __restrict__ ehat,
    const __bf16* __restrict__ W1t, const float* __restrict__ b1,
    const __bf16* __restrict__ W2t, const float* __restrict__ b2,
    const float* __restrict__ W3, const float* __restrict__ b3,
    const float* __restrict__ bscal,
    float* __restrict__ Ypred, float* __restrict__ PW) {
  extern __shared__ char smem[];
  __bf16* W1L = (__bf16*)smem;            // 256 x 264
  __bf16* W2L = W1L + HH * WLP;           // 256 x 264
  __bf16* zA  = W2L + HH * WLP;           // 32 x 264
  __bf16* zB  = zA + KK * LDP;            // 32 x 264
  float*  b1L = (float*)(zB + KK * LDP);  // 256
  float*  b2L = b1L + HH;                 // 256
  float*  W3L = b2L + HH;                 // 256
  float*  mlp_s = W3L + HH;               // 32
  int*    nbr_s = (int*)(mlp_s + KK);     // 32

  const int tid = threadIdx.x;
  const int lane = tid & 31;
  const int wave = tid >> 5;

  // ---- stage weights into LDS once (amortized over ~NN/GRID nodes)
  for (int c = tid; c < 8192; c += 256) {       // 8192 x 8 bf16 = 256x256
    const int row = c >> 5;
    const int off = (c & 31) * 8;
    *(bf16x8*)(W1L + (size_t)row * WLP + off) =
        *(const bf16x8*)(W1t + (size_t)row * HH + off);
    *(bf16x8*)(W2L + (size_t)row * WLP + off) =
        *(const bf16x8*)(W2t + (size_t)row * HH + off);
  }
  if (tid < HH) {
    b1L[tid] = b1[tid];
    b2L[tid] = b2[tid];
    W3L[tid] = W3[tid];
  }
  const float bb  = bscal[0];
  const float b3v = b3[0];
  __syncthreads();

  for (int i = blockIdx.x; i < NN; i += gridDim.x) {
    // prefetch next node's index row into cache (global_prefetch_b8)
    const int inext = i + gridDim.x;
    if (inext < NN) __builtin_prefetch(nbrs + (size_t)inext * KK, 0, 1);

    if (tid < KK) nbr_s[tid] = nbrs[(size_t)i * KK + tid];
    __syncthreads();

    // build z_concat: row k = [ x[i] (128) | x[nbr_k] (128) ], converted to bf16
    const float* xi = x + (size_t)i * DD;
    for (int e = tid; e < KK * 2 * DD; e += 256) {
      const int k = e >> 8;
      const int c = e & 255;
      float v = (c < DD) ? xi[c] : x[(size_t)nbr_s[k] * DD + (c - DD)];
      zA[(size_t)k * LDP + c] = (__bf16)v;
    }
    __syncthreads();

    dense_layer_wmma(zA, zB, W1L, b1L, lane, wave);   // h1 = relu(z @ W1 + b1)
    __syncthreads();
    dense_layer_wmma(zB, zA, W2L, b2L, lane, wave);   // h2 = relu(h1 @ W2 + b2)
    __syncthreads();

    // layer 3: mlp[k] = h2[k,:] . W3 + b3   (8 threads per row, 32 cols each)
    {
      const int r = tid >> 3;
      const int sub = tid & 7;
      const int c0 = sub * 32;
      float s = 0.0f;
#pragma unroll
      for (int c = 0; c < 32; ++c)
        s += (float)zA[(size_t)r * LDP + c0 + c] * W3L[c0 + c];
      s += __shfl_xor(s, 1);
      s += __shfl_xor(s, 2);
      s += __shfl_xor(s, 4);
      if (sub == 0) mlp_s[r] = s + b3v;
    }
    __syncthreads();

    // softmax over K=32 + outputs, wave 0 (one lane per neighbor)
    if (wave == 0) {
      const float m = mlp_s[lane];
      const float a = bb * fabsf(m);
      float mx = a;
#pragma unroll
      for (int off = 16; off > 0; off >>= 1) mx = fmaxf(mx, __shfl_xor(mx, off));
      const float ex = __expf(a - mx);
      float sum = ex;
#pragma unroll
      for (int off = 16; off > 0; off >>= 1) sum += __shfl_xor(sum, off);
      const float pwv = m * (ex / sum);

      const int idx = nbr_s[lane];
      const float pr = t[idx] - ehat[idx];
      float yv = pr * pwv;
#pragma unroll
      for (int off = 16; off > 0; off >>= 1) yv += __shfl_xor(yv, off);
      if (lane == 0) Ypred[i] = yv;

      PW[(size_t)nbr_s[0] * NN + idx] = pwv;   // row cur[i] == i
    }
    __syncthreads();   // nbr_s / zA must not be overwritten until wave 0 is done
  }
}

extern "C" void kernel_launch(void* const* d_in, const int* in_sizes, int n_in,
                              void* d_out, int out_size, void* d_ws, size_t ws_size,
                              hipStream_t stream) {
  (void)in_sizes; (void)n_in; (void)out_size; (void)ws_size;
  const float* x    = (const float*)d_in[0];
  const int*   nbrs = (const int*)d_in[1];
  const float* t    = (const float*)d_in[2];
  const float* ehat = (const float*)d_in[3];
  const float* W1   = (const float*)d_in[4];
  const float* b1   = (const float*)d_in[5];
  const float* W2   = (const float*)d_in[6];
  const float* b2   = (const float*)d_in[7];
  const float* W3   = (const float*)d_in[8];
  const float* b3   = (const float*)d_in[9];
  const float* b    = (const float*)d_in[10];

  float* Ypred = (float*)d_out;             // N floats
  float* PW    = (float*)d_out + NN;        // N*N floats

  __bf16* W1t = (__bf16*)d_ws;              // 256x256 bf16 = 128KB
  __bf16* W2t = W1t + (size_t)HH * HH;      // next 128KB

  // allow > default dynamic LDS (300.3 KB of the 320 KB per-WGP pool)
  hipFuncSetAttribute((const void*)gcn_attn_kernel,
                      hipFuncAttributeMaxDynamicSharedMemorySize, SMEM_BYTES);

  prep_weights<<<256, 256, 0, stream>>>(W1, W2, W1t, W2t);
  zero_out<<<4096, 256, 0, stream>>>(PW, (size_t)NN * NN / 4);
  gcn_attn_kernel<<<GRID, 256, SMEM_BYTES, stream>>>(x, nbrs, t, ehat, W1t, b1,
                                                     W2t, b2, W3, b3, b,
                                                     Ypred, PW);
}
